// Encoder_6150393168190
// MI455X (gfx1250) — compile-verified, matrix operand order checked by reading, not run on hardware
//
#include <hip/hip_runtime.h>
#include <hip/hip_bf16.h>

typedef _Float16 f16;
typedef __attribute__((ext_vector_type(8)))  _Float16 v8h;
typedef __attribute__((ext_vector_type(16))) _Float16 v16h;
typedef __attribute__((ext_vector_type(8)))  float    v8f;

#define EMB      64
#define IN_DIM   11
#define EDGE_DIM 4
#define LAT      32
#define NN       50000
#define NE       800000
#define NG       512
#define BN_EPS   1e-5f

// ---------------- WMMA helpers ----------------
// A layout (16-bit A 16x32): lane L<16 -> row M=L, halves = K[0..7],K[16..23];
// lane L+16 -> same row, halves = K[8..15],K[24..31].
__device__ __forceinline__ v16h load_a16(const f16* p) {
  v8h lo = *(const v8h*)(p);
  v8h hi = *(const v8h*)(p + 16);
  v16h a;
#pragma unroll
  for (int j = 0; j < 8; ++j) { a[j] = lo[j]; a[8 + j] = hi[j]; }
  return a;
}

// Register-resident-B WMMA GEMM: Y[rows, ncols] = A[rows, NCH*32](f16) * W + bias.
// One wave per block; B fragments (NCH K-chunks x COLT col-tiles) are loaded into
// VGPRs once, then the wave iterates TPW 16-row tiles, so steady state is
// A-loads + WMMAs only. Next tile's A row is prefetched (global_prefetch).
template <int COLT, int NCH, int TPW>
__global__ void gemm_reg(const f16* __restrict__ A, const f16* __restrict__ Wp,
                         const float* __restrict__ bias, float* __restrict__ Y,
                         int ncols, int ntiles) {
  const int lane = threadIdx.x;
  const int lrow = lane & 15, lhi = lane >> 4;
  const int K = NCH * 32;
  v16h b[NCH][COLT];
#pragma unroll
  for (int c = 0; c < NCH; ++c)
#pragma unroll
    for (int t = 0; t < COLT; ++t)
      b[c][t] = *(const v16h*)(Wp + (((long)(t * NCH + c) * 32 + lane) << 4));
  float bz[COLT];
#pragma unroll
  for (int t = 0; t < COLT; ++t) bz[t] = bias[t * 16 + lrow];

  const int tile0 = blockIdx.x * TPW;
#pragma unroll 1
  for (int it = 0; it < TPW; ++it) {
    const int tile = tile0 + it;
    if (tile >= ntiles) break;
    const long row0 = (long)tile * 16;
    const f16* arow = A + (row0 + lrow) * (long)K;
    if (it + 1 < TPW && tile + 1 < ntiles)
      __builtin_prefetch(arow + 16 * (long)K, 0, 1); // next tile's A row
    v8f acc[COLT];
#pragma unroll
    for (int t = 0; t < COLT; ++t) acc[t] = {};
#pragma unroll
    for (int c = 0; c < NCH; ++c) {
      v16h a = load_a16(arow + c * 32 + 8 * lhi);
#pragma unroll
      for (int t = 0; t < COLT; ++t)
        acc[t] = __builtin_amdgcn_wmma_f32_16x16x32_f16(false, a, false, b[c][t],
                                                        (short)0, acc[t], false, false);
    }
#pragma unroll
    for (int t = 0; t < COLT; ++t) {
      int col = t * 16 + lrow;
#pragma unroll
      for (int r = 0; r < 8; ++r)
        Y[(row0 + r + 8 * lhi) * ncols + col] = acc[t][r] + bz[t];
    }
  }
}

// Message GEMM 1 with fused gather: A row e = [h16[dst[e]](64) | h16[src[e]](64) |
// edge_attr(4) | dist(1) | 0-pad] (K=160). Avoids materializing 256 MB concat.
// B reloaded per chunk (holding 160 B VGPRs would hurt occupancy where random
// gather latency hiding matters most); weights stay hot in WGP$/L2.
__global__ void msg1_gemm(const f16* __restrict__ h16, const int* __restrict__ src,
                          const int* __restrict__ dst, const float* __restrict__ eattr,
                          const float* __restrict__ eaux, const f16* __restrict__ Wp,
                          const float* __restrict__ bias, float* __restrict__ Y) {
  const int lane = threadIdx.x;
  const int lrow = lane & 15, lhi = lane >> 4;
  const long row0 = (long)blockIdx.x * 16;
  const long e = row0 + lrow;
  const int s = src[e];
  const int d = dst[e];
  v8f acc[4];
#pragma unroll
  for (int t = 0; t < 4; ++t) acc[t] = {};
#pragma unroll
  for (int c = 0; c < 5; ++c) {
    v16h a;
    if (c < 4) {
      const long node = (c < 2) ? (long)d : (long)s;
      a = load_a16(h16 + node * EMB + (c & 1) * 32 + 8 * lhi);
    } else {
#pragma unroll
      for (int j = 0; j < 16; ++j) {
        int q = (j < 8) ? (8 * lhi + j) : (16 + 8 * lhi + (j - 8)); // K-128
        float v = 0.f;
        if (q < 4) v = eattr[e * 4 + q];
        else if (q == 4) v = eaux[e * 4 + 3]; // dist
        a[j] = (f16)v;
      }
    }
#pragma unroll
    for (int t = 0; t < 4; ++t) {
      v16h b = *(const v16h*)(Wp + (((long)(t * 5 + c) * 32 + lane) << 4));
      acc[t] = __builtin_amdgcn_wmma_f32_16x16x32_f16(false, a, false, b,
                                                      (short)0, acc[t], false, false);
    }
  }
#pragma unroll
  for (int t = 0; t < 4; ++t) {
    int col = t * 16 + lrow;
    float bz = bias[col];
#pragma unroll
    for (int r = 0; r < 8; ++r) Y[(row0 + r + 8 * lhi) * EMB + col] = acc[t][r] + bz;
  }
}

// Pack W[K_real, ncols] (f32 row-major) into per-lane WMMA B layout (f16),
// zero-padding K to a multiple of 32.
__global__ void pack_w(const float* __restrict__ W, f16* __restrict__ Wp,
                       int nchunks, int Kreal, int ncols, int total) {
  int idx = blockIdx.x * blockDim.x + threadIdx.x;
  if (idx >= total) return;
  int lane = idx & 31;
  int c = (idx >> 5) % nchunks;
  int t = idx / (32 * nchunks);
  int n = t * 16 + (lane & 15);
  int lhi = lane >> 4;
#pragma unroll
  for (int j = 0; j < 16; ++j) {
    int kk = c * 32 + ((j < 8) ? (8 * lhi + j) : (16 + 8 * lhi + (j - 8)));
    float v = (kk < Kreal) ? W[(long)kk * ncols + n] : 0.f;
    Wp[((long)idx << 4) + j] = (f16)v;
  }
}

// ---------------- elementwise / reduction kernels ----------------
__global__ void zero_f32(float* p, long n) {
  long i = (long)blockIdx.x * blockDim.x + threadIdx.x;
  if (i < n) p[i] = 0.f;
}
__global__ void copy_f32(const float* a, float* b, long n) {
  long i = (long)blockIdx.x * blockDim.x + threadIdx.x;
  if (i < n) b[i] = a[i];
}
__global__ void f32_to_f16(const float* a, f16* b, long n) {
  long i = (long)blockIdx.x * blockDim.x + threadIdx.x;
  if (i < n) b[i] = (f16)a[i];
}
__global__ void build_xpad(const float* __restrict__ x, f16* __restrict__ xp, long n) {
  long i = (long)blockIdx.x * blockDim.x + threadIdx.x;
  if (i >= n) return;
  long row = i >> 5; int c = (int)(i & 31);
  xp[i] = (c < IN_DIM) ? (f16)x[row * IN_DIM + c] : (f16)0.f;
}
__global__ void edge_geom(const float* __restrict__ pos, const int* __restrict__ src,
                          const int* __restrict__ dst, float* __restrict__ eaux, long E) {
  long e = (long)blockIdx.x * blockDim.x + threadIdx.x;
  if (e >= E) return;
  long s = src[e], d = dst[e];
  float dx = pos[d * 3 + 0] - pos[s * 3 + 0];
  float dy = pos[d * 3 + 1] - pos[s * 3 + 1];
  float dz = pos[d * 3 + 2] - pos[s * 3 + 2];
  eaux[e * 4 + 0] = dx; eaux[e * 4 + 1] = dy; eaux[e * 4 + 2] = dz;
  eaux[e * 4 + 3] = sqrtf(dx * dx + dy * dy + dz * dz);
}
// Per-column (64) sum & sumsq -> stats[0:64]=sum, stats[64:128]=sumsq
__global__ void colstats(const float* __restrict__ Y, long rows, float* __restrict__ stats) {
  int c = threadIdx.x; // 64 threads
  float s = 0.f, q = 0.f;
  for (long r = blockIdx.x; r < rows; r += gridDim.x) {
    float v = Y[r * EMB + c];
    s += v; q += v * v;
  }
  atomicAdd(&stats[c], s);
  atomicAdd(&stats[64 + c], q);
}
__global__ void bn_finalize(float* stats, const float* g, const float* b, float n) {
  int c = threadIdx.x; // 64
  float mean = stats[c] / n;
  float var = stats[64 + c] / n - mean * mean;
  float sc = g[c] * rsqrtf(var + BN_EPS);
  stats[128 + c] = sc;
  stats[192 + c] = b[c] - sc * mean;
}
__global__ void bn_relu16(const float* __restrict__ Y, const float* __restrict__ stats,
                          f16* __restrict__ out, long n) {
  long i = (long)blockIdx.x * blockDim.x + threadIdx.x;
  if (i >= n) return;
  int c = (int)(i & 63);
  float v = stats[128 + c] * Y[i] + stats[192 + c];
  out[i] = (f16)fmaxf(v, 0.f);
}
__global__ void bn_relu32(const float* __restrict__ Y, const float* __restrict__ stats,
                          float* __restrict__ out, long n) {
  long i = (long)blockIdx.x * blockDim.x + threadIdx.x;
  if (i >= n) return;
  int c = (int)(i & 63);
  out[i] = fmaxf(stats[128 + c] * Y[i] + stats[192 + c], 0.f);
}
__global__ void wdot64(const f16* __restrict__ T, const float* __restrict__ W,
                       const float* __restrict__ b, float* __restrict__ w, long E) {
  long e = (long)blockIdx.x * blockDim.x + threadIdx.x;
  if (e >= E) return;
  float s = b[0];
#pragma unroll
  for (int c = 0; c < EMB; ++c) s += (float)T[e * EMB + c] * W[c];
  w[e] = s;
}
__global__ void scatter_edges(const f16* __restrict__ m16, const int* __restrict__ dst,
                              const float* __restrict__ eaux, const float* __restrict__ w,
                              float* __restrict__ magg, float* __restrict__ psum,
                              float* __restrict__ cnt, long E) {
  long e = (long)blockIdx.x * blockDim.x + threadIdx.x;
  if (e >= E) return;
  long d = dst[e];
  float we = w[e];
#pragma unroll
  for (int c = 0; c < EMB; ++c) atomicAdd(&magg[d * EMB + c], (float)m16[e * EMB + c]);
#pragma unroll
  for (int k = 0; k < 3; ++k) atomicAdd(&psum[d * 3 + k], eaux[e * 4 + k] * we);
  atomicAdd(&cnt[d], 1.f);
}
__global__ void build_uin(const f16* __restrict__ h16, const float* __restrict__ magg,
                          f16* __restrict__ uin, long n) {
  long i = (long)blockIdx.x * blockDim.x + threadIdx.x;
  if (i >= n) return;
  long node = i >> 7; int c = (int)(i & 127);
  uin[i] = (c < EMB) ? h16[node * EMB + c] : (f16)magg[node * EMB + (c - EMB)];
}
__global__ void add_h(float* __restrict__ h, const float* __restrict__ u,
                      f16* __restrict__ h16, long n) {
  long i = (long)blockIdx.x * blockDim.x + threadIdx.x;
  if (i >= n) return;
  float v = h[i] + u[i];
  h[i] = v; h16[i] = (f16)v;
}
__global__ void pos_update(float* __restrict__ pos, const float* __restrict__ psum,
                           const float* __restrict__ cnt, long N) {
  long nidx = (long)blockIdx.x * blockDim.x + threadIdx.x;
  if (nidx >= N) return;
  float c = fmaxf(cnt[nidx], 1.f);
#pragma unroll
  for (int k = 0; k < 3; ++k) pos[nidx * 3 + k] += psum[nidx * 3 + k] / c;
}
__global__ void pool(const float* __restrict__ h, const int* __restrict__ batch,
                     float* __restrict__ hs, float* __restrict__ gcnt, long N) {
  long nidx = (long)blockIdx.x * blockDim.x + threadIdx.x;
  if (nidx >= N) return;
  long b = batch[nidx];
#pragma unroll
  for (int c = 0; c < EMB; ++c) atomicAdd(&hs[b * EMB + c], h[nidx * EMB + c]);
  atomicAdd(&gcnt[b], 1.f);
}
__global__ void hgraph(const float* __restrict__ hs, const float* __restrict__ gcnt,
                       float* __restrict__ hg, f16* __restrict__ hg16, long n) {
  long i = (long)blockIdx.x * blockDim.x + threadIdx.x;
  if (i >= n) return;
  float v = hs[i] / fmaxf(gcnt[i >> 6], 1.f);
  hg[i] = v; hg16[i] = (f16)v;
}
__global__ void propdot(const f16* __restrict__ z16, const float* __restrict__ W,
                        const float* __restrict__ b, float* __restrict__ out, int G) {
  int g = blockIdx.x * blockDim.x + threadIdx.x;
  if (g >= G) return;
  float s = b[0];
#pragma unroll
  for (int c = 0; c < EMB; ++c) s += (float)z16[(long)g * EMB + c] * W[c];
  out[g] = s;
}

// ---------------- host launch ----------------
// params pytree leaf order (JAX sorts dict keys): layers[0], layers[1],
// lin_in, log_var, mu, pp_bn, pp_l1, pp_l2; _lin -> (W, b); _bn -> (b, g).
#define PBASE 3
#define LSTRIDE 22
enum { MSG_BN1_B=0, MSG_BN1_G, MSG_BN2_B, MSG_BN2_G, MSG_L1_W, MSG_L1_B,
       MSG_L2_W, MSG_L2_B, POS_BN_B, POS_BN_G, POS_L1_W, POS_L1_B,
       POS_L2_W, POS_L2_B, UPD_BN1_B, UPD_BN1_G, UPD_BN2_B, UPD_BN2_G,
       UPD_L1_W, UPD_L1_B, UPD_L2_W, UPD_L2_B };
#define LIN_IN_W 47
#define LIN_IN_B 48
#define LOGVAR_W 49
#define LOGVAR_B 50
#define MU_W     51
#define MU_B     52
#define PP_BN_B  53
#define PP_BN_G  54
#define PP_L1_W  55
#define PP_L1_B  56
#define PP_L2_W  57
#define PP_L2_B  58
#define EDGE_IDX 59
#define BATCH_IX 60

static inline long cdiv(long a, long b) { return (a + b - 1) / b; }

extern "C" void kernel_launch(void* const* d_in, const int* in_sizes, int n_in,
                              void* d_out, int out_size, void* d_ws, size_t ws_size,
                              hipStream_t stream) {
  (void)in_sizes; (void)n_in; (void)out_size; (void)ws_size;
  const float* x_in   = (const float*)d_in[0];
  const float* pos_in = (const float*)d_in[1];
  const float* eattr  = (const float*)d_in[2];
  const int*   eidx   = (const int*)d_in[EDGE_IDX];
  const int*   src    = eidx;            // edge_index[0]
  const int*   dstn   = eidx + NE;       // edge_index[1]
  const int*   batch  = (const int*)d_in[BATCH_IX];
  float* out = (float*)d_out;
  auto P = [&](int i) { return (const float*)d_in[i]; };

  char* ws = (char*)d_ws; size_t off = 0;
  auto alloc = [&](size_t bytes) -> void* {
    void* p = ws + off; off = (off + bytes + 255) & ~(size_t)255; return p;
  };
  float* posb  = (float*)alloc(NN * 3 * 4);
  float* h     = (float*)alloc((size_t)NN * EMB * 4);
  f16*   h16   = (f16*)alloc((size_t)NN * EMB * 2);
  f16*   xpad  = (f16*)alloc((size_t)NN * 32 * 2);
  float* eaux  = (float*)alloc((size_t)NE * 4 * 4);
  float* Ey    = (float*)alloc((size_t)NE * EMB * 4);
  f16*   e16A  = (f16*)alloc((size_t)NE * EMB * 2);
  f16*   e16B  = (f16*)alloc((size_t)NE * EMB * 2);
  float* wbuf  = (float*)alloc((size_t)NE * 4);
  float* magg  = (float*)alloc((size_t)NN * EMB * 4);
  float* psum  = (float*)alloc(NN * 3 * 4);
  float* cnt   = (float*)alloc(NN * 4);
  f16*   uin16 = (f16*)alloc((size_t)NN * 128 * 2);
  float* uy    = (float*)alloc((size_t)NN * EMB * 4);
  f16*   u16   = (f16*)alloc((size_t)NN * EMB * 2);
  float* ub    = (float*)alloc((size_t)NN * EMB * 4);
  float* stats = (float*)alloc(256 * 4);
  float* hs    = (float*)alloc((size_t)NG * EMB * 4);
  float* gcnt  = (float*)alloc(NG * 4);
  float* hg    = (float*)alloc((size_t)NG * EMB * 4);
  f16*   hg16  = (f16*)alloc((size_t)NG * EMB * 2);
  f16*   mu16  = (f16*)alloc((size_t)NG * LAT * 2);
  float* zy    = (float*)alloc((size_t)NG * EMB * 4);
  f16*   z16   = (f16*)alloc((size_t)NG * EMB * 2);
  f16* wp_in   = (f16*)alloc(4 * 1 * 32 * 16 * 2);
  f16* wp_msg1 = (f16*)alloc(4 * 5 * 32 * 16 * 2);
  f16* wp_msg2 = (f16*)alloc(4 * 2 * 32 * 16 * 2);
  f16* wp_pos1 = (f16*)alloc(4 * 2 * 32 * 16 * 2);
  f16* wp_upd1 = (f16*)alloc(4 * 4 * 32 * 16 * 2);
  f16* wp_upd2 = (f16*)alloc(4 * 2 * 32 * 16 * 2);
  f16* wp_mu   = (f16*)alloc(2 * 2 * 32 * 16 * 2);
  f16* wp_lv   = (f16*)alloc(2 * 2 * 32 * 16 * 2);
  f16* wp_pp1  = (f16*)alloc(4 * 1 * 32 * 16 * 2);

  const int T = 256;
  const int TPW = 8; // row tiles per wave in register-resident-B GEMMs
  const int ET = NE / 16, NT = NN / 16, GT = NG / 16; // 16-row tile counts
  auto pack = [&](const float* W, f16* Wp, int K, int Kreal, int ncols) {
    int total = (ncols / 16) * (K / 32) * 32;
    pack_w<<<(int)cdiv(total, T), T, 0, stream>>>(W, Wp, K / 32, Kreal, ncols, total);
  };
  auto bnpass16 = [&](const float* Y, long rows, const float* g, const float* b, f16* o) {
    zero_f32<<<1, 128, 0, stream>>>(stats, 128);
    colstats<<<1024, 64, 0, stream>>>(Y, rows, stats);
    bn_finalize<<<1, 64, 0, stream>>>(stats, g, b, (float)rows);
    bn_relu16<<<(int)cdiv(rows * EMB, T), T, 0, stream>>>(Y, stats, o, rows * EMB);
  };

  // lin_in
  copy_f32<<<(int)cdiv(NN * 3, T), T, 0, stream>>>(pos_in, posb, NN * 3);
  build_xpad<<<(int)cdiv((long)NN * 32, T), T, 0, stream>>>(x_in, xpad, (long)NN * 32);
  pack(P(LIN_IN_W), wp_in, 32, IN_DIM, EMB);
  gemm_reg<4, 1, 8><<<(int)cdiv(NT, TPW), 32, 0, stream>>>(xpad, wp_in, P(LIN_IN_B), h,
                                                           EMB, NT);
  f32_to_f16<<<(int)cdiv((long)NN * EMB, T), T, 0, stream>>>(h, h16, (long)NN * EMB);

  for (int L = 0; L < 2; ++L) {
    const int B = PBASE + L * LSTRIDE;
    pack(P(B + MSG_L1_W), wp_msg1, 160, 2 * EMB + EDGE_DIM + 1, EMB);
    pack(P(B + MSG_L2_W), wp_msg2, 64, EMB, EMB);
    pack(P(B + POS_L1_W), wp_pos1, 64, EMB, EMB);
    pack(P(B + UPD_L1_W), wp_upd1, 128, 2 * EMB, EMB);
    pack(P(B + UPD_L2_W), wp_upd2, 64, EMB, EMB);

    edge_geom<<<(int)cdiv(NE, T), T, 0, stream>>>(posb, src, dstn, eaux, NE);
    // msg MLP
    msg1_gemm<<<ET, 32, 0, stream>>>(h16, src, dstn, eattr, eaux, wp_msg1,
                                     P(B + MSG_L1_B), Ey);
    bnpass16(Ey, NE, P(B + MSG_BN1_G), P(B + MSG_BN1_B), e16A);
    gemm_reg<4, 2, 8><<<(int)cdiv(ET, TPW), 32, 0, stream>>>(e16A, wp_msg2,
                                                             P(B + MSG_L2_B), Ey, EMB, ET);
    bnpass16(Ey, NE, P(B + MSG_BN2_G), P(B + MSG_BN2_B), e16B); // m
    // pos MLP
    gemm_reg<4, 2, 8><<<(int)cdiv(ET, TPW), 32, 0, stream>>>(e16B, wp_pos1,
                                                             P(B + POS_L1_B), Ey, EMB, ET);
    bnpass16(Ey, NE, P(B + POS_BN_G), P(B + POS_BN_B), e16A);
    wdot64<<<(int)cdiv(NE, T), T, 0, stream>>>(e16A, P(B + POS_L2_W), P(B + POS_L2_B), wbuf, NE);
    // aggregate
    zero_f32<<<(int)cdiv((long)NN * EMB, T), T, 0, stream>>>(magg, (long)NN * EMB);
    zero_f32<<<(int)cdiv(NN * 3, T), T, 0, stream>>>(psum, NN * 3);
    zero_f32<<<(int)cdiv(NN, T), T, 0, stream>>>(cnt, NN);
    scatter_edges<<<(int)cdiv(NE, T), T, 0, stream>>>(e16B, dstn, eaux, wbuf,
                                                      magg, psum, cnt, NE);
    // update MLP
    build_uin<<<(int)cdiv((long)NN * 128, T), T, 0, stream>>>(h16, magg, uin16, (long)NN * 128);
    gemm_reg<4, 4, 8><<<(int)cdiv(NT, TPW), 32, 0, stream>>>(uin16, wp_upd1,
                                                             P(B + UPD_L1_B), uy, EMB, NT);
    bnpass16(uy, NN, P(B + UPD_BN1_G), P(B + UPD_BN1_B), u16);
    gemm_reg<4, 2, 8><<<(int)cdiv(NT, TPW), 32, 0, stream>>>(u16, wp_upd2,
                                                             P(B + UPD_L2_B), uy, EMB, NT);
    zero_f32<<<1, 128, 0, stream>>>(stats, 128);
    colstats<<<1024, 64, 0, stream>>>(uy, NN, stats);
    bn_finalize<<<1, 64, 0, stream>>>(stats, P(B + UPD_BN2_G), P(B + UPD_BN2_B), (float)NN);
    bn_relu32<<<(int)cdiv((long)NN * EMB, T), T, 0, stream>>>(uy, stats, ub, (long)NN * EMB);
    add_h<<<(int)cdiv((long)NN * EMB, T), T, 0, stream>>>(h, ub, h16, (long)NN * EMB);
    pos_update<<<(int)cdiv(NN, T), T, 0, stream>>>(posb, psum, cnt, NN);
  }

  // global mean pool + heads
  zero_f32<<<(int)cdiv((long)NG * EMB, T), T, 0, stream>>>(hs, (long)NG * EMB);
  zero_f32<<<(int)cdiv(NG, T), T, 0, stream>>>(gcnt, NG);
  pool<<<(int)cdiv(NN, T), T, 0, stream>>>(h, batch, hs, gcnt, NN);
  hgraph<<<(int)cdiv((long)NG * EMB, T), T, 0, stream>>>(hs, gcnt, hg, hg16, (long)NG * EMB);

  pack(P(MU_W), wp_mu, 64, EMB, LAT);
  pack(P(LOGVAR_W), wp_lv, 64, EMB, LAT);
  gemm_reg<2, 2, 8><<<(int)cdiv(GT, TPW), 32, 0, stream>>>(hg16, wp_mu, P(MU_B), out,
                                                           LAT, GT);
  gemm_reg<2, 2, 8><<<(int)cdiv(GT, TPW), 32, 0, stream>>>(hg16, wp_lv, P(LOGVAR_B),
                                                           out + NG * LAT, LAT, GT);

  f32_to_f16<<<(int)cdiv((long)NG * LAT, T), T, 0, stream>>>(out, mu16, (long)NG * LAT);
  pack(P(PP_L1_W), wp_pp1, 32, LAT, EMB);
  gemm_reg<4, 1, 8><<<(int)cdiv(GT, TPW), 32, 0, stream>>>(mu16, wp_pp1, P(PP_L1_B),
                                                           zy, EMB, GT);
  zero_f32<<<1, 128, 0, stream>>>(stats, 128);
  colstats<<<512, 64, 0, stream>>>(zy, NG, stats);
  bn_finalize<<<1, 64, 0, stream>>>(stats, P(PP_BN_G), P(PP_BN_B), (float)NG);
  bn_relu16<<<(int)cdiv((long)NG * EMB, T), T, 0, stream>>>(zy, stats, z16, (long)NG * EMB);
  propdot<<<cdiv(NG, T), T, 0, stream>>>(z16, P(PP_L2_W), P(PP_L2_B), out + 2 * NG * LAT, NG);
}
